// Attention_Head_67147518705845
// MI455X (gfx1250) — matrix operand order repaired and checked by reading, not compiled, not run on hardware
//
#include <hip/hip_runtime.h>
#include <math.h>

typedef __attribute__((ext_vector_type(16))) _Float16 v16h;
typedef __attribute__((ext_vector_type(8)))  _Float16 v8h;
typedef __attribute__((ext_vector_type(8)))  float    v8f;

#define BATCH 4
#define SEQ   4096
#define DMODEL 1024
#define HDIM  128
#define SCALE 0.08838834764831845f  // 1/sqrt(128)
#define NKT   (SEQ / 32)            // 128 key-tile iterations

static __device__ __forceinline__ v8f wmma_f16(v16h a, v16h b, v8f c) {
  return __builtin_amdgcn_wmma_f32_16x16x32_f16(false, a, false, b, (short)0, c,
                                                false, false);
}

static __device__ __forceinline__ v16h cat8(v8h lo, v8h hi) {
  return __builtin_shufflevector(lo, hi, 0, 1, 2, 3, 4, 5, 6, 7, 8, 9, 10, 11,
                                 12, 13, 14, 15);
}

// ---------------------------------------------------------------------------
// Kernel 0: convert + transpose weights: Wt[h][d] = (f16) W[d][h]
// ---------------------------------------------------------------------------
__global__ __launch_bounds__(256) void ah_prep_kernel(
    const float* __restrict__ Wq, const float* __restrict__ Wk,
    const float* __restrict__ Wv, _Float16* __restrict__ wtq,
    _Float16* __restrict__ wtk, _Float16* __restrict__ wtv) {
  int o = blockIdx.x * blockDim.x + threadIdx.x;   // 0 .. D*H-1
  int h = o / DMODEL;
  int d = o - h * DMODEL;
  int src = d * HDIM + h;
  wtq[o] = (_Float16)Wq[src];
  wtk[o] = (_Float16)Wk[src];
  wtv[o] = (_Float16)Wv[src];
}

// ---------------------------------------------------------------------------
// Kernel 1: projections. One block (8 waves) owns one 16-row strip of x.
// The 16x32 x chunk is staged f32->f16 into LDS once per K-step (double-
// buffered) and shared by all 8 waves (one wave per 16-wide H column tile).
//   Q  f16 [b][s][h], Kh f16 [b][s][h], Vt f16 [b][h][s]
// ---------------------------------------------------------------------------
__global__ __launch_bounds__(256) void ah_proj_kernel(
    const float* __restrict__ x, const _Float16* __restrict__ wtq,
    const _Float16* __restrict__ wtk, const _Float16* __restrict__ wtv,
    _Float16* __restrict__ Q, _Float16* __restrict__ Kh,
    _Float16* __restrict__ Vt) {
  __shared__ __align__(32) _Float16 xbuf[2][16 * 32];

  const int t    = threadIdx.x;
  const int wid  = t >> 5;          // = H column tile (0..7)
  const int lane = t & 31;
  const int sel  = lane >> 4;
  const int n    = lane & 15;

  const int rt    = blockIdx.x;                    // 1024 row strips
  const int b     = rt >> 8;
  const int sbase = (rt & 255) << 4;

  // staging assignment: thread loads 2 consecutive floats of the 16x32 chunk
  const int xr = t >> 4;            // row 0..15
  const int xc = (t & 15) * 2;      // col 0..30 step 2
  const float* xg = x + ((size_t)(b * SEQ + sbase + xr)) * DMODEL + xc;

  v8f cq = {}, ck = {}, cv = {};
  const size_t wb = (size_t)(wid * 16 + n) * DMODEL + sel * 16;

  float2 xf = *(const float2*)xg;   // chunk 0
  for (int kc = 0; kc < DMODEL / 32; ++kc) {
    const int cur = kc & 1;
    xbuf[cur][xr * 32 + xc]     = (_Float16)xf.x;
    xbuf[cur][xr * 32 + xc + 1] = (_Float16)xf.y;
    __syncthreads();
    if (kc + 1 < DMODEL / 32) xf = *(const float2*)(xg + (kc + 1) * 32);

    // ---- A fragment from LDS (broadcast to all 8 waves) ----
    v16h a = *(const v16h*)&xbuf[cur][n * 32 + sel * 16];

    // ---- B fragments: Wt[col][d], contraction d contiguous ----
    v16h bq = *(const v16h*)(wtq + wb + kc * 32);
    v16h bk = *(const v16h*)(wtk + wb + kc * 32);
    v16h bv = *(const v16h*)(wtv + wb + kc * 32);
    cq = wmma_f16(a, bq, cq);
    ck = wmma_f16(a, bk, ck);
    cv = wmma_f16(a, bv, cv);
  }

  // ---- store tiles (C layout: vgpr r -> row M = r + 8*sel, col N = n) ----
  const int hbase = wid * 16;
#pragma unroll
  for (int r = 0; r < 8; ++r) {
    int M = r + 8 * sel;
    size_t qi = ((size_t)(b * SEQ + sbase + M)) * HDIM + hbase + n;
    Q[qi]  = (_Float16)cq[r];
    Kh[qi] = (_Float16)ck[r];
    Vt[((size_t)(b * HDIM + hbase + n)) * SEQ + sbase + M] = (_Float16)cv[r];
  }
}

// ---------------------------------------------------------------------------
// Kernel 2: fused flash attention. One block (8 waves) shares double-
// buffered K/V tiles in LDS; one wave owns one 16-query tile.
// ---------------------------------------------------------------------------
__global__ __launch_bounds__(256) void ah_attn_kernel(
    const _Float16* __restrict__ Q, const _Float16* __restrict__ Kh,
    const _Float16* __restrict__ Vt, float* __restrict__ out) {
  __shared__ __align__(32) _Float16 kbuf[2][32 * HDIM];  // [key][h]
  __shared__ __align__(32) _Float16 vbuf[2][HDIM * 32];  // [h][key]
  __shared__ __align__(32) _Float16 pbuf[8][16 * 32];    // per-wave P tile

  const int t    = threadIdx.x;
  const int wid  = t >> 5;
  const int lane = t & 31;
  const int sel  = lane >> 4;
  const int n    = lane & 15;

  const int qt    = blockIdx.x * 8 + wid;   // 1024 query tiles; same b per block
  const int b     = qt >> 8;
  const int qbase = (qt & 255) << 4;

  // staging assignment: each thread moves one 16-half segment of K and of V
  const int krow = t >> 3, kseg = t & 7;    // K tile: 32 rows x 8 segs
  const int vrow = t >> 1, vseg = t & 1;    // V tile: 128 rows x 2 segs
  const _Float16* kg =
      Kh + ((size_t)(b * SEQ) + krow) * HDIM + kseg * 16;      // + kt*32*HDIM
  const _Float16* vg =
      Vt + ((size_t)(b * HDIM) + vrow) * SEQ + vseg * 16;      // + kt*32

  // ---- load Q A-fragments for all 4 head-dim chunks (kept resident) ----
  v16h aq[4];
  const _Float16* qrow = Q + ((size_t)(b * SEQ + qbase + n)) * HDIM;
#pragma unroll
  for (int hc = 0; hc < 4; ++hc) {
    aq[hc] = *(const v16h*)(qrow + hc * 32 + sel * 16);
  }

  v8f co[8];
  float m_r[8], l_r[8];
#pragma unroll
  for (int i = 0; i < 8; ++i) {
    co[i] = {};
    m_r[i] = -1.0e30f;
    l_r[i] = 0.0f;
  }

  v16h kreg = *(const v16h*)kg;   // tile 0
  v16h vreg = *(const v16h*)vg;

  for (int kt = 0; kt < NKT; ++kt) {
    const int cur = kt & 1;
    *(v16h*)&kbuf[cur][krow * HDIM + kseg * 16] = kreg;
    *(v16h*)&vbuf[cur][vrow * 32 + vseg * 16]   = vreg;
    __syncthreads();
    // issue next tile's global loads now; consumed after compute
    if (kt + 1 < NKT) {
      kreg = *(const v16h*)(kg + (size_t)(kt + 1) * 32 * HDIM);
      vreg = *(const v16h*)(vg + (kt + 1) * 32);
    }
    if (kt + 2 < NKT) {
      __builtin_prefetch(kg + (size_t)(kt + 2) * 32 * HDIM, 0, 0);
      __builtin_prefetch(vg + (kt + 2) * 32, 0, 0);
    }

    // ---- scores: two 16-key tiles, contraction over h (from LDS) ----
    v8f cs0 = {}, cs1 = {};
#pragma unroll
    for (int hc = 0; hc < 4; ++hc) {
      v16h b0 = *(const v16h*)&kbuf[cur][n * HDIM + hc * 32 + sel * 16];
      v16h b1 = *(const v16h*)&kbuf[cur][(16 + n) * HDIM + hc * 32 + sel * 16];
      cs0 = wmma_f16(aq[hc], b0, cs0);
      cs1 = wmma_f16(aq[hc], b1, cs1);
    }

    // ---- online softmax per row (rows of this half live in vgpr r) ----
    float alpha[8];
#pragma unroll
    for (int r = 0; r < 8; ++r) {
      float s0 = cs0[r] * SCALE;
      float s1 = cs1[r] * SCALE;
      float v = fmaxf(s0, s1);
      v = fmaxf(v, __shfl_xor(v, 1));
      v = fmaxf(v, __shfl_xor(v, 2));
      v = fmaxf(v, __shfl_xor(v, 4));
      v = fmaxf(v, __shfl_xor(v, 8));
      float mnew = fmaxf(m_r[r], v);
      alpha[r] = __expf(m_r[r] - mnew);
      m_r[r] = mnew;
      float p0 = __expf(s0 - mnew);
      float p1 = __expf(s1 - mnew);
      float ps = p0 + p1;
      ps += __shfl_xor(ps, 1);
      ps += __shfl_xor(ps, 2);
      ps += __shfl_xor(ps, 4);
      ps += __shfl_xor(ps, 8);
      l_r[r] = l_r[r] * alpha[r] + ps;
      // stage P (C layout -> row-major 16x32 in per-wave LDS)
      int M = r + 8 * sel;
      pbuf[wid][M * 32 + n]      = (_Float16)p0;
      pbuf[wid][M * 32 + 16 + n] = (_Float16)p1;
    }

    // ---- rescale output accumulators ----
#pragma unroll
    for (int nt = 0; nt < 8; ++nt)
#pragma unroll
      for (int r = 0; r < 8; ++r) co[nt][r] *= alpha[r];

    // ---- reload P as A-fragment (row = n, A 16x32 layout) ----
    const _Float16* pb = &pbuf[wid][n * 32];
    v16h ap = *(const v16h*)(pb + sel * 16);

    // ---- P @ V : 8 output column tiles, contraction over 32 keys ----
#pragma unroll
    for (int nt = 0; nt < 8; ++nt) {
      v16h bv = *(const v16h*)&vbuf[cur][(nt * 16 + n) * 32 + sel * 16];
      co[nt] = wmma_f16(ap, bv, co[nt]);
    }
  }

  // ---- normalize + store f32 output ----
#pragma unroll
  for (int r = 0; r < 8; ++r) {
    int M = r + 8 * sel;
    float inv = 1.0f / l_r[r];
    size_t ob = ((size_t)(b * SEQ + qbase + M)) * HDIM + n;
#pragma unroll
    for (int nt = 0; nt < 8; ++nt) out[ob + nt * 16] = co[nt][r] * inv;
  }
}

// ---------------------------------------------------------------------------
extern "C" void kernel_launch(void* const* d_in, const int* in_sizes, int n_in,
                              void* d_out, int out_size, void* d_ws,
                              size_t ws_size, hipStream_t stream) {
  const float* x  = (const float*)d_in[0];
  const float* Wq = (const float*)d_in[1];
  const float* Wk = (const float*)d_in[2];
  const float* Wv = (const float*)d_in[3];
  float* out = (float*)d_out;

  const size_t WSZ = (size_t)DMODEL * HDIM;          // 131072
  const size_t QSZ = (size_t)BATCH * SEQ * HDIM;     // 2097152
  _Float16* ws  = (_Float16*)d_ws;
  _Float16* wtq = ws;
  _Float16* wtk = wtq + WSZ;
  _Float16* wtv = wtk + WSZ;
  _Float16* Qh  = wtv + WSZ;
  _Float16* Kh  = Qh + QSZ;
  _Float16* Vt  = Kh + QSZ;

  ah_prep_kernel<<<WSZ / 256, 256, 0, stream>>>(Wq, Wk, Wv, wtq, wtk, wtv);
  ah_proj_kernel<<<1024, 256, 0, stream>>>(x, wtq, wtk, wtv, Qh, Kh, Vt);
  ah_attn_kernel<<<128, 256, 0, stream>>>(Qh, Kh, Vt, out);
}
